// GPSTransformer_70729521430710
// MI455X (gfx1250) — matrix operand order repaired and checked by reading, not compiled, not run on hardware
//
#include <hip/hip_runtime.h>
#include <hip/hip_bf16.h>
#include <cstdint>
#include <cstddef>

// ---------------------------------------------------------------------------
// GPS graph-transformer forward for gfx1250 (MI455X), wave32 + WMMA f16.
// GEMMs: v_wmma_f32_16x16x32_f16, double-buffered LDS tiles staged with
// GLOBAL_LOAD_ASYNC_TO_LDS_B128 (async DMA path, tracked by ASYNCcnt).
// ---------------------------------------------------------------------------

typedef __attribute__((ext_vector_type(16))) _Float16 v16h;
typedef __attribute__((ext_vector_type(8)))  float    v8f;
typedef int v4i_g __attribute__((vector_size(16)));   // GCC-vector int4 (builtin V4i)

#if __has_builtin(__builtin_amdgcn_global_load_async_to_lds_b128)
#define USE_ASYNC_LDS 1
#else
#define USE_ASYNC_LDS 0
#endif

// builtin signature: (v4i addrspace(1)*, v4i addrspace(3)*, imm offset, imm cpol)
#define ASG(p) ((__attribute__((address_space(1))) v4i_g*)(p))
#define ASL(p) ((__attribute__((address_space(3))) v4i_g*)(p))

__device__ __forceinline__ void wait_async_lds() {
#if USE_ASYNC_LDS
#if __has_builtin(__builtin_amdgcn_s_wait_asynccnt)
  __builtin_amdgcn_s_wait_asynccnt(0);
#else
  asm volatile("s_wait_asynccnt 0" ::: "memory");
#endif
#endif
}

__device__ __forceinline__ v8f wmma_f16(v16h a, v16h b, v8f c) {
  return __builtin_amdgcn_wmma_f32_16x16x32_f16(
      /*neg_a=*/false, a, /*neg_b=*/false, b,
      /*c_mod=*/(short)0, c, /*reuse_a=*/false, /*reuse_b=*/false);
}

// A-fragment (16x32, 16-bit): lanes 0-15 -> row r, K 0..7 & 16..23;
// lanes 16-31 -> row r, K 8..15 & 24..31.  (ISA 7.12.2, 16-bit A 16x32)
__device__ __forceinline__ v16h frag_a(const _Float16* base, int ld, int lane) {
  int r = lane & 15, hf = lane >> 4;
  const _Float16* p = base + r * ld + hf * 8;
  union { uint4 u[2]; v16h v; } f;
  f.u[0] = *(const uint4*)(p);
  f.u[1] = *(const uint4*)(p + 16);
  return f.v;
}

// B-fragment (32x16, 16-bit), operand stored transposed [N x K] row-major:
// lane = column (lane&15); lanes 0-15 -> K 0..7 & 16..23, lanes 16-31 -> K 8..15 & 24..31.
__device__ __forceinline__ v16h frag_b(const _Float16* base, int ld, int lane) {
  int c = lane & 15, hf = lane >> 4;
  const _Float16* p = base + c * ld + hf * 8;
  union { uint4 u[2]; v16h v; } f;
  f.u[0] = *(const uint4*)(p);
  f.u[1] = *(const uint4*)(p + 16);
  return f.v;
}

// ---------------------------------------------------------------------------
// WMMA GEMM: out = act(A[MxK]_f16 @ Bt[NxK]_f16^T + bias) (+ res).
// 128x128 block tile, 8 waves, 64x32 per wave, double-buffered K tiles.
// ACT: 0 none, 1 relu, 2 leaky.  Compile-time flags kill epilogue branches.
// ---------------------------------------------------------------------------
template <int ACT, bool RES, bool OF, bool OH>
__global__ __launch_bounds__(256) void k_gemm(
    const _Float16* __restrict__ A, const _Float16* __restrict__ Bt,
    const float* __restrict__ bias, const float* __restrict__ res,
    float* __restrict__ outF, _Float16* __restrict__ outH,
    int Nc, int K)
{
  __shared__ __align__(16) _Float16 As[2][128][40];
  __shared__ __align__(16) _Float16 Bs[2][128][40];
  const int tid = threadIdx.x, lane = tid & 31, wave = tid >> 5;
  const int bm = blockIdx.y * 128, bn = blockIdx.x * 128;
  const int wm = (wave & 1) * 64, wn = (wave >> 1) * 32;

  v8f acc[4][2] = {};

  const int row = tid >> 1, hf = tid & 1;
  const _Float16* gA = A  + (size_t)(bm + row) * K + hf * 16;
  const _Float16* gB = Bt + (size_t)(bn + row) * K + hf * 16;

  auto stage = [&](int buf, int k0) {
#if USE_ASYNC_LDS
    __builtin_amdgcn_global_load_async_to_lds_b128(
        ASG(gA + k0),     ASL(&As[buf][row][hf * 16 + 0]), 0, 0);
    __builtin_amdgcn_global_load_async_to_lds_b128(
        ASG(gA + k0 + 8), ASL(&As[buf][row][hf * 16 + 8]), 0, 0);
    __builtin_amdgcn_global_load_async_to_lds_b128(
        ASG(gB + k0),     ASL(&Bs[buf][row][hf * 16 + 0]), 0, 0);
    __builtin_amdgcn_global_load_async_to_lds_b128(
        ASG(gB + k0 + 8), ASL(&Bs[buf][row][hf * 16 + 8]), 0, 0);
#else
    uint4 a0 = ((const uint4*)(gA + k0))[0];
    uint4 a1 = ((const uint4*)(gA + k0))[1];
    uint4 b0 = ((const uint4*)(gB + k0))[0];
    uint4 b1 = ((const uint4*)(gB + k0))[1];
    *(uint4*)&As[buf][row][hf * 16 + 0] = a0;
    *(uint4*)&As[buf][row][hf * 16 + 8] = a1;
    *(uint4*)&Bs[buf][row][hf * 16 + 0] = b0;
    *(uint4*)&Bs[buf][row][hf * 16 + 8] = b1;
#endif
  };

  stage(0, 0);
  wait_async_lds();
  __syncthreads();

  int cur = 0;
  for (int k0 = 0; k0 < K; k0 += 32) {
    if (k0 + 32 < K) stage(cur ^ 1, k0 + 32);

    v16h af[4], bf[2];
#pragma unroll
    for (int i = 0; i < 4; ++i) af[i] = frag_a(&As[cur][wm + i * 16][0], 40, lane);
#pragma unroll
    for (int j = 0; j < 2; ++j) bf[j] = frag_b(&Bs[cur][wn + j * 16][0], 40, lane);
#pragma unroll
    for (int i = 0; i < 4; ++i)
#pragma unroll
      for (int j = 0; j < 2; ++j)
        acc[i][j] = wmma_f16(af[i], bf[j], acc[i][j]);

    wait_async_lds();
    __syncthreads();
    cur ^= 1;
  }

  const int cb = lane & 15, rh = lane >> 4;   // C/D layout: VGPR e -> row (8*half+e)
#pragma unroll
  for (int i = 0; i < 4; ++i) {
#pragma unroll
    for (int j = 0; j < 2; ++j) {
      int col = bn + wn + j * 16 + cb;
      int rb  = bm + wm + i * 16 + rh * 8;
#pragma unroll
      for (int e = 0; e < 8; ++e) {
        int r = rb + e;
        float v = acc[i][j][e] + bias[col];
        if (ACT == 1)      v = fmaxf(v, 0.f);
        else if (ACT == 2) v = (v >= 0.f) ? v : 0.01f * v;
        if (RES) v += res[(size_t)r * Nc + col];
        if (OF)  outF[(size_t)r * Nc + col] = v;
        if (OH)  outH[(size_t)r * Nc + col] = (_Float16)v;
      }
    }
  }
}

// ---------------------------------------------------------------------------
// Fused per-(graph, head) attention: S=256 keys, d=32.  K and V^T live in
// LDS; scores + P@V via WMMA; 8-lane-group shuffle softmax.
// ---------------------------------------------------------------------------
__global__ __launch_bounds__(256) void k_attn(const float* __restrict__ qkv,
                                              _Float16* __restrict__ o16)
{
  __shared__ __align__(16) _Float16 Ks[256][40];   // [key][d]
  __shared__ __align__(16) _Float16 Vt[32][264];   // [d][key]
  __shared__ __align__(16) _Float16 Qs[32][40];    // [row][d]
  __shared__ __align__(16) _Float16 Sp[32][264];   // scores then P (f16)
  const float scale = 0.17677669529663687f;        // 1/sqrt(32)
  const int tid = threadIdx.x, lane = tid & 31, wave = tid >> 5;
  const int g = blockIdx.x >> 3, hd = blockIdx.x & 7;
  const int nodeBase = g * 256;

  for (int i = tid; i < 256 * 32; i += 256) {
    int s = i >> 5, d = i & 31;
    Ks[s][d] = (_Float16)qkv[(size_t)(nodeBase + s) * 768 + 256 + hd * 32 + d];
    Vt[d][s] = (_Float16)qkv[(size_t)(nodeBase + s) * 768 + 512 + hd * 32 + d];
  }
  __syncthreads();

  for (int rb = 0; rb < 8; ++rb) {
    const int row0 = rb * 32;
    for (int i = tid; i < 32 * 32; i += 256) {
      int s = i >> 5, d = i & 31;
      Qs[s][d] = (_Float16)qkv[(size_t)(nodeBase + row0 + s) * 768 + hd * 32 + d];
    }
    __syncthreads();

    { // scores: 32x256 = 2 row-tiles x 16 col-tiles, 4 tiles per wave
      int rt = wave & 1, cbt = (wave >> 1) * 4;
      v16h aq = frag_a(&Qs[rt * 16][0], 40, lane);
#pragma unroll
      for (int j = 0; j < 4; ++j) {
        v16h bk = frag_b(&Ks[(cbt + j) * 16][0], 40, lane);
        v8f sc = {};
        sc = wmma_f16(aq, bk, sc);
        int col  = (cbt + j) * 16 + (lane & 15);
        int rloc = rt * 16 + (lane >> 4) * 8;
#pragma unroll
        for (int e = 0; e < 8; ++e)
          Sp[rloc + e][col] = (_Float16)(sc[e] * scale);
      }
    }
    __syncthreads();

    { // softmax: 8 lanes per row, each owns 32 cols
      int r = tid >> 3, sub = tid & 7;
      float vals[32], mx = -3.0e38f;
#pragma unroll
      for (int k = 0; k < 32; ++k) {
        vals[k] = (float)Sp[r][sub * 32 + k];
        mx = fmaxf(mx, vals[k]);
      }
      for (int m = 1; m < 8; m <<= 1) mx = fmaxf(mx, __shfl_xor(mx, m, 32));
      float sum = 0.f;
#pragma unroll
      for (int k = 0; k < 32; ++k) { vals[k] = __expf(vals[k] - mx); sum += vals[k]; }
      for (int m = 1; m < 8; m <<= 1) sum += __shfl_xor(sum, m, 32);
      float inv = 1.f / sum;
#pragma unroll
      for (int k = 0; k < 32; ++k) Sp[r][sub * 32 + k] = (_Float16)(vals[k] * inv);
    }
    __syncthreads();

    if (wave < 4) { // O = P(32x256) @ V(256x32): 4 tiles, K loop of 8
      int rt = wave & 1, ct = wave >> 1;
      v8f acc = {};
#pragma unroll
      for (int kk = 0; kk < 8; ++kk) {
        v16h a = frag_a(&Sp[rt * 16][kk * 32], 264, lane);
        v16h b = frag_b(&Vt[ct * 16][kk * 32], 264, lane);
        acc = wmma_f16(a, b, acc);
      }
      int d    = ct * 16 + (lane & 15);
      int rloc = rt * 16 + (lane >> 4) * 8;
#pragma unroll
      for (int e = 0; e < 8; ++e)
        o16[(size_t)(nodeBase + row0 + rloc + e) * 256 + hd * 32 + d] = (_Float16)acc[e];
    }
    __syncthreads();
  }
}

// ---------------------------------------------------------------------------
// GINE edge messages: agg[dst] += relu(h[src] + edge_attr @ We + be)
// ---------------------------------------------------------------------------
__global__ __launch_bounds__(256) void k_edge(
    const float* __restrict__ h, const float* __restrict__ ea,
    const int* __restrict__ src, const int* __restrict__ dst,
    const float* __restrict__ We, const float* __restrict__ be,
    float* __restrict__ agg)
{
  int e = blockIdx.x, c = threadIdx.x;
  float a0 = ea[e * 2 + 0], a1 = ea[e * 2 + 1];
  float ec = a0 * We[c] + a1 * We[256 + c] + be[c];
  float m = h[(size_t)src[e] * 256 + c] + ec;
  m = fmaxf(m, 0.f);
  atomicAdd(&agg[(size_t)dst[e] * 256 + c], m);
}

// Column stats for BatchNorm: one thread per column, 256 rows per block.
__global__ __launch_bounds__(256) void k_colstats(
    const float* __restrict__ in, int ncols,
    float* __restrict__ sum, float* __restrict__ sumsq)
{
  int c = threadIdx.x;
  if (c >= ncols) return;
  size_t r0 = (size_t)blockIdx.x * 256;
  float s = 0.f, s2 = 0.f;
  for (int r = 0; r < 256; ++r) {
    float v = in[(r0 + r) * ncols + c];
    s += v; s2 += v * v;
  }
  atomicAdd(&sum[c], s);
  atomicAdd(&sumsq[c], s2);
}

__global__ __launch_bounds__(256) void k_bn(
    const float* __restrict__ in,
    const float* __restrict__ sum, const float* __restrict__ sumsq,
    const float* __restrict__ gam, const float* __restrict__ bet,
    float* __restrict__ outF, _Float16* __restrict__ outH, int ncols)
{
  size_t idx = (size_t)blockIdx.x * 256 + threadIdx.x;
  int c = (int)(idx % ncols);
  const float invN = 1.0f / 16384.f;
  float mu = sum[c] * invN;
  float var = sumsq[c] * invN - mu * mu;
  float rstd = rsqrtf(var + 1e-5f);
  float v = (in[idx] - mu) * rstd * gam[c] + bet[c];
  if (outF) outF[idx] = v;
  if (outH) outH[idx] = (_Float16)v;
}

__global__ __launch_bounds__(256) void k_addcvt(
    const float* __restrict__ a, const float* __restrict__ b,
    _Float16* __restrict__ oh, float* __restrict__ of)
{
  size_t i = (size_t)blockIdx.x * 256 + threadIdx.x;
  float v = a[i] + b[i];
  if (of) of[i] = v;
  oh[i] = (_Float16)v;
}

__global__ __launch_bounds__(256) void k_cvt(const float* __restrict__ a,
                                             _Float16* __restrict__ oh)
{
  size_t i = (size_t)blockIdx.x * 256 + threadIdx.x;
  oh[i] = (_Float16)a[i];
}

// Encoder: t = leaky(x @ enc_W + enc_b), IN=9 -> ENC=236
__global__ __launch_bounds__(256) void k_enc(
    const float* __restrict__ x, const float* __restrict__ W,
    const float* __restrict__ b, float* __restrict__ t)
{
  int n = blockIdx.x, c = threadIdx.x;
  if (c >= 236) return;
  float acc = b[c];
#pragma unroll
  for (int i = 0; i < 9; ++i) acc += x[n * 9 + i] * W[i * 236 + c];
  acc = (acc >= 0.f) ? acc : 0.01f * acc;
  t[(size_t)n * 236 + c] = acc;
}

// h = concat(bn(t), bn(pe))
__global__ __launch_bounds__(256) void k_concat(
    const float* __restrict__ t, const float* __restrict__ pe,
    const float* __restrict__ s1, const float* __restrict__ q1,
    const float* __restrict__ g1, const float* __restrict__ b1,
    const float* __restrict__ s2, const float* __restrict__ q2,
    const float* __restrict__ g2, const float* __restrict__ b2,
    float* __restrict__ h)
{
  int n = blockIdx.x, c = threadIdx.x;
  const float invN = 1.0f / 16384.f;
  float v;
  if (c < 236) {
    float mu = s1[c] * invN;
    float var = q1[c] * invN - mu * mu;
    float rstd = rsqrtf(var + 1e-5f);
    v = (t[(size_t)n * 236 + c] - mu) * rstd * g1[c] + b1[c];
  } else {
    int cp = c - 236;
    float mu = s2[cp] * invN;
    float var = q2[cp] * invN - mu * mu;
    float rstd = rsqrtf(var + 1e-5f);
    v = (pe[(size_t)n * 20 + cp] - mu) * rstd * g2[cp] + b2[cp];
  }
  h[(size_t)n * 256 + c] = v;
}

// Weight prep: f32 -> f16 (already [out,K]) and transpose+convert ([K,out] -> [out,K])
__global__ __launch_bounds__(256) void k_wcvt(const float* __restrict__ w,
                                              _Float16* __restrict__ o, int n)
{
  int i = blockIdx.x * 256 + threadIdx.x;
  if (i < n) o[i] = (_Float16)w[i];
}

__global__ __launch_bounds__(256) void k_wtr(const float* __restrict__ w,
                                             _Float16* __restrict__ o, int K, int Nout)
{
  int i = blockIdx.x * 256 + threadIdx.x;
  if (i < K * Nout) {
    int k = i / Nout, n = i % Nout;
    o[(size_t)n * K + k] = (_Float16)w[i];
  }
}

// Final tiny projection to OUT=6
__global__ __launch_bounds__(256) void k_dec2(
    const _Float16* __restrict__ d1, const float* __restrict__ W,
    const float* __restrict__ b, float* __restrict__ out)
{
  int i = blockIdx.x * 256 + threadIdx.x;  // N*6
  int n = i / 6, o = i % 6;
  float acc = b[o];
  for (int k = 0; k < 256; ++k)
    acc += (float)d1[(size_t)n * 256 + k] * W[k * 6 + o];
  out[i] = acc;
}

// ---------------------------------------------------------------------------
extern "C" void kernel_launch(void* const* d_in, const int* in_sizes, int n_in,
                              void* d_out, int out_size, void* d_ws, size_t ws_size,
                              hipStream_t stream)
{
  (void)in_sizes; (void)n_in; (void)out_size; (void)ws_size;
  constexpr int N = 16384, C = 256, G = 64, H = 8, L = 8, E = 131072;
  constexpr int ENC = 236, PEc = 20, OUT = 6;

  const float* x     = (const float*)d_in[0];
  const float* pe    = (const float*)d_in[1];
  const int*   ei    = (const int*)  d_in[2];
  const float* eattr = (const float*)d_in[3];
  const float* encW  = (const float*)d_in[5];
  const float* encB  = (const float*)d_in[6];
  const float* in_g  = (const float*)d_in[7];
  const float* in_b  = (const float*)d_in[8];
  const float* pe_g  = (const float*)d_in[9];
  const float* pe_b  = (const float*)d_in[10];
  const float* gWe   = (const float*)d_in[11];
  const float* gbe   = (const float*)d_in[12];
  const float* gWn   = (const float*)d_in[13];
  const float* gbn   = (const float*)d_in[14];
  const float* Wqkv  = (const float*)d_in[15];
  const float* bqkv  = (const float*)d_in[16];
  const float* Wo    = (const float*)d_in[17];
  const float* bo    = (const float*)d_in[18];
  const float* W1    = (const float*)d_in[19];
  const float* b1    = (const float*)d_in[20];
  const float* W2    = (const float*)d_in[21];
  const float* b2    = (const float*)d_in[22];
  const float* bn_g  = (const float*)d_in[23];
  const float* bn_b  = (const float*)d_in[24];
  const float* pre_g = (const float*)d_in[25];
  const float* pre_b = (const float*)d_in[26];
  const float* dW1   = (const float*)d_in[27];
  const float* db1   = (const float*)d_in[28];
  const float* dW2   = (const float*)d_in[29];
  const float* db2   = (const float*)d_in[30];
  const int* src = ei;
  const int* dst = ei + E;

  char* p = (char*)d_ws;
  auto alloc = [&](size_t bytes) {
    void* r = (void*)p;
    p += (bytes + 255) & ~(size_t)255;
    return r;
  };
  float*    h    = (float*)alloc((size_t)N * C * 4);
  float*    fA   = (float*)alloc((size_t)N * C * 4);
  float*    fB   = (float*)alloc((size_t)N * C * 4);
  float*    fC   = (float*)alloc((size_t)N * C * 4);
  float*    fD   = (float*)alloc((size_t)N * C * 4);
  float*    qkv  = (float*)alloc((size_t)N * 3 * C * 4);
  _Float16* h16  = (_Float16*)alloc((size_t)N * C * 2);
  _Float16* t16  = (_Float16*)alloc((size_t)N * C * 2);
  _Float16* o16  = (_Float16*)alloc((size_t)N * C * 2);
  _Float16* m16  = (_Float16*)alloc((size_t)N * 2 * C * 2);
  _Float16* d16  = (_Float16*)alloc((size_t)N * C * 2);
  float* stats = (float*)alloc(4096 * 4);
  float *ssum = stats, *ssq = stats + 1024, *ssum2 = stats + 2048, *ssq2 = stats + 3072;
  _Float16* wqkv16 = (_Float16*)alloc((size_t)3 * C * C * 2);
  _Float16* wo16   = (_Float16*)alloc((size_t)C * C * 2);
  _Float16* wnT16  = (_Float16*)alloc((size_t)C * C * 2);
  _Float16* w1T16  = (_Float16*)alloc((size_t)C * 2 * C * 2);
  _Float16* w2T16  = (_Float16*)alloc((size_t)2 * C * C * 2);
  _Float16* wd1T16 = (_Float16*)alloc((size_t)C * C * 2);

  const int TPB = 256;
  const int NC_BLK = N * C / TPB;

  // ---- encoder ----
  k_enc<<<N, TPB, 0, stream>>>(x, encW, encB, fA);
  (void)hipMemsetAsync(stats, 0, 4096 * 4, stream);
  k_colstats<<<64, TPB, 0, stream>>>(fA, ENC, ssum, ssq);
  k_colstats<<<64, TPB, 0, stream>>>(pe, PEc, ssum2, ssq2);
  k_concat<<<N, TPB, 0, stream>>>(fA, pe, ssum, ssq, in_g, in_b,
                                  ssum2, ssq2, pe_g, pe_b, h);

  // ---- layers ----
  for (int l = 0; l < L; ++l) {
    const float* We_l   = gWe  + (size_t)l * 2 * C;
    const float* be_l   = gbe  + (size_t)l * C;
    const float* Wn_l   = gWn  + (size_t)l * C * C;
    const float* bnn_l  = gbn  + (size_t)l * C;
    const float* Wqkv_l = Wqkv + (size_t)l * 3 * C * C;
    const float* bqkv_l = bqkv + (size_t)l * 3 * C;
    const float* Wo_l   = Wo   + (size_t)l * C * C;
    const float* bo_l   = bo   + (size_t)l * C;
    const float* W1_l   = W1   + (size_t)l * C * 2 * C;
    const float* b1_l   = b1   + (size_t)l * 2 * C;
    const float* W2_l   = W2   + (size_t)l * 2 * C * C;
    const float* b2_l   = b2   + (size_t)l * C;
    const float* bng    = bn_g + (size_t)l * 4 * C;
    const float* bnb    = bn_b + (size_t)l * 4 * C;

    // device-side weight prep (f32 -> f16, output-major)
    k_wtr <<<C * C / TPB,     TPB, 0, stream>>>(Wn_l,   wnT16, C, C);
    k_wcvt<<<3 * C * C / TPB, TPB, 0, stream>>>(Wqkv_l, wqkv16, 3 * C * C);
    k_wcvt<<<C * C / TPB,     TPB, 0, stream>>>(Wo_l,   wo16,   C * C);
    k_wtr <<<C * 2 * C / TPB, TPB, 0, stream>>>(W1_l,   w1T16, C, 2 * C);
    k_wtr <<<2 * C * C / TPB, TPB, 0, stream>>>(W2_l,   w2T16, 2 * C, C);

    // GINE conv
    (void)hipMemsetAsync(fB, 0, (size_t)N * C * 4, stream);
    k_edge<<<E, TPB, 0, stream>>>(h, eattr, src, dst, We_l, be_l, fB);
    k_addcvt<<<NC_BLK, TPB, 0, stream>>>(h, fB, t16, nullptr);   // t16 = f16(h+agg)
    k_gemm<2, true, true, false><<<dim3(C / 128, N / 128), TPB, 0, stream>>>(
        t16, wnT16, bnn_l, h, fC, nullptr, C, C);
    (void)hipMemsetAsync(stats, 0, 2048 * 4, stream);
    k_colstats<<<64, TPB, 0, stream>>>(fC, C, ssum, ssq);
    k_bn<<<NC_BLK, TPB, 0, stream>>>(fC, ssum, ssq, bng, bnb, fD, nullptr, C); // hloc_n

    // global attention
    k_cvt<<<NC_BLK, TPB, 0, stream>>>(h, h16);
    k_gemm<0, false, true, false><<<dim3(3 * C / 128, N / 128), TPB, 0, stream>>>(
        h16, wqkv16, bqkv_l, nullptr, qkv, nullptr, 3 * C, C);
    k_attn<<<G * H, TPB, 0, stream>>>(qkv, o16);
    k_gemm<0, true, true, false><<<dim3(C / 128, N / 128), TPB, 0, stream>>>(
        o16, wo16, bo_l, h, fB, nullptr, C, C);
    (void)hipMemsetAsync(stats, 0, 2048 * 4, stream);
    k_colstats<<<64, TPB, 0, stream>>>(fB, C, ssum, ssq);
    k_bn<<<NC_BLK, TPB, 0, stream>>>(fB, ssum, ssq, bng + C, bnb + C, fC, nullptr, C); // hatt_n

    // combine + FFN
    k_addcvt<<<NC_BLK, TPB, 0, stream>>>(fD, fC, t16, fB);       // out = hloc_n + hatt_n
    k_gemm<1, false, false, true><<<dim3(2 * C / 128, N / 128), TPB, 0, stream>>>(
        t16, w1T16, b1_l, nullptr, nullptr, m16, 2 * C, C);
    k_gemm<0, true, true, false><<<dim3(C / 128, N / 128), TPB, 0, stream>>>(
        m16, w2T16, b2_l, fB, fC, nullptr, C, 2 * C);
    (void)hipMemsetAsync(stats, 0, 2048 * 4, stream);
    k_colstats<<<64, TPB, 0, stream>>>(fC, C, ssum, ssq);
    k_bn<<<NC_BLK, TPB, 0, stream>>>(fC, ssum, ssq, bng + 2 * C, bnb + 2 * C,
                                     fD, nullptr, C);            // norm3
    (void)hipMemsetAsync(stats, 0, 2048 * 4, stream);
    k_colstats<<<64, TPB, 0, stream>>>(fD, C, ssum, ssq);
    k_bn<<<NC_BLK, TPB, 0, stream>>>(fD, ssum, ssq, bng + 3 * C, bnb + 3 * C,
                                     h, nullptr, C);             // outer BN -> h
  }

  // ---- decoder ----
  k_wtr<<<C * C / TPB, TPB, 0, stream>>>(dW1, wd1T16, C, C);
  (void)hipMemsetAsync(stats, 0, 2048 * 4, stream);
  k_colstats<<<64, TPB, 0, stream>>>(h, C, ssum, ssq);
  k_bn<<<NC_BLK, TPB, 0, stream>>>(h, ssum, ssq, pre_g, pre_b, nullptr, t16, C);
  k_gemm<2, false, false, true><<<dim3(C / 128, N / 128), TPB, 0, stream>>>(
      t16, wd1T16, db1, nullptr, nullptr, d16, C, C);
  k_dec2<<<N * OUT / TPB, TPB, 0, stream>>>(d16, dW2, db2, (float*)d_out);
}
